// SwinTransformerBlock_74569222193821
// MI455X (gfx1250) — matrix operand order repaired and checked by reading, not compiled, not run on hardware
//
#include <hip/hip_runtime.h>
#include <hip/hip_bf16.h>

// ---------------------------------------------------------------------------
// Swin block for MI455X (gfx1250): bf16 WMMA (v_wmma_f32_16x16x32_bf16)
// + async global->LDS tile staging (ASYNCcnt path) when the toolchain has it.
// ---------------------------------------------------------------------------

#define WIN   32
#define SHIFT 16
#define CDIM  512
#define NTOK  4096
#define BDIM  8
#define HDIM  2048
#define NWIN  128
#define MTOK  (BDIM * NTOK)   // 32768 tokens

typedef __attribute__((ext_vector_type(16))) __bf16 v16bf;
typedef __attribute__((ext_vector_type(8)))  float  v8f;
typedef __attribute__((ext_vector_type(4)))  int    v4i;

#if defined(__has_builtin)
#if __has_builtin(__builtin_amdgcn_global_load_async_to_lds_b128)
#define ASYNC_LDS 1
#endif
#endif

// Stage 16 bytes global -> LDS. Async path uses GLOBAL_LOAD_ASYNC_TO_LDS_B128
// (no VGPR round-trip, tracked by ASYNCcnt); fallback is load+ds_store.
static __device__ inline void stage16(const __bf16* g, __bf16* l) {
#if ASYNC_LDS
    __builtin_amdgcn_global_load_async_to_lds_b128(
        (__attribute__((address_space(1))) v4i*)g,
        (__attribute__((address_space(3))) v4i*)l, 0, 0);
#else
    *(uint4*)l = *(const uint4*)g;
#endif
}

// Wait for this wave's outstanding async LDS writes (then caller barriers).
static __device__ inline void async_join() {
#if ASYNC_LDS
#if __has_builtin(__builtin_amdgcn_s_wait_asynccnt)
    __builtin_amdgcn_s_wait_asynccnt(0);
#else
    asm volatile("s_wait_asynccnt 0x0" ::: "memory");
#endif
#endif
}

union Frag16 {
    v16bf  v;
    uint4  u[2];
    __bf16 e[16];
};

static __device__ inline v8f zero8() {
    v8f z;
#pragma unroll
    for (int i = 0; i < 8; ++i) z[i] = 0.0f;
    return z;
}

// A-matrix 16x32 bf16 fragment: lanes 0-15 hold K=0..15 consecutive;
// lanes 16-31 hold K=16..31 (row M = lane%16). base = row start + k0.
static __device__ inline v16bf load_frag_a(const __bf16* base, int hi) {
    Frag16 f;
    const int o = hi ? 16 : 0;
    f.u[0] = *(const uint4*)(base + o);
    f.u[1] = *(const uint4*)(base + o + 8);
    return f.v;
}

// B-matrix 32x16 bf16 fragment: lane holds column N=lane%16; lanes 0-15 hold
// K=k0..k0+15 consecutive, lanes 16-31 hold K=k0+16..k0+31 (cf. sparse B
// layout ISA 7.12.4). base = K-contiguous source row for this lane's column.
static __device__ inline v16bf load_frag_bk(const __bf16* base, int hi) {
    Frag16 f;
    const int o = hi ? 16 : 0;
    f.u[0] = *(const uint4*)(base + o);
    f.u[1] = *(const uint4*)(base + o + 8);
    return f.v;
}

static __device__ inline v8f wmma_bf16(v16bf a, v16bf b, v8f c) {
    return __builtin_amdgcn_wmma_f32_16x16x32_bf16(
        /*neg_a=*/false, a, /*neg_b=*/false, b,
        /*c_mod=*/(short)0, c, /*reuse_a=*/false, /*reuse_b=*/false);
}

static __device__ inline float gelu_exact(float x) {
    return 0.5f * x * (1.0f + erff(x * 0.70710678118654752f));
}

// ---------------------------------------------------------------------------
// fp32 -> bf16 conversion (weights)
// ---------------------------------------------------------------------------
__global__ __launch_bounds__(256)
void cvt_bf16_kernel(const float* __restrict__ src, __bf16* __restrict__ dst, int n) {
    int i = blockIdx.x * 256 + threadIdx.x;
    if (i < n) dst[i] = (__bf16)src[i];
}

// ---------------------------------------------------------------------------
// LayerNorm over C=512, output bf16. One block (128 thr) per token row.
// ---------------------------------------------------------------------------
__global__ __launch_bounds__(128)
void layernorm_bf16_kernel(const float* __restrict__ x, __bf16* __restrict__ y,
                           const float* __restrict__ w, const float* __restrict__ b) {
    const int row  = blockIdx.x;
    const int tid  = threadIdx.x;
    const int wave = tid >> 5, lane = tid & 31;
    const float* xr = x + (long long)row * CDIM;

    float vals[4];
    float s = 0.0f, s2 = 0.0f;
#pragma unroll
    for (int i = 0; i < 4; ++i) {
        float v = xr[tid + i * 128];
        vals[i] = v; s += v; s2 += v * v;
    }
#pragma unroll
    for (int o = 16; o > 0; o >>= 1) {
        s  += __shfl_xor(s,  o, 32);
        s2 += __shfl_xor(s2, o, 32);
    }
    __shared__ float rs[4], rs2[4];
    if (lane == 0) { rs[wave] = s; rs2[wave] = s2; }
    __syncthreads();
    s  = rs[0] + rs[1] + rs[2] + rs[3];
    s2 = rs2[0] + rs2[1] + rs2[2] + rs2[3];
    const float mu   = s * (1.0f / CDIM);
    const float var  = s2 * (1.0f / CDIM) - mu * mu;
    const float rstd = rsqrtf(var + 1e-5f);
#pragma unroll
    for (int i = 0; i < 4; ++i) {
        int c = tid + i * 128;
        y[(long long)row * CDIM + c] = (__bf16)((vals[i] - mu) * rstd * w[c] + b[c]);
    }
}

// ---------------------------------------------------------------------------
// Windowed attention. One block (4 waves) per (batch, window).
// Shift applied via token indexing; output scatter implements the
// "swapaxes(1,2).reshape" scramble + roll(+SHIFT) + residual, writing kv_out.
// ---------------------------------------------------------------------------
__global__ __launch_bounds__(128)
void attn_kernel(const __bf16* __restrict__ qn, const __bf16* __restrict__ kn,
                 const float* __restrict__ kv, const float* __restrict__ mask,
                 float* __restrict__ kv_out) {
    const int bw   = blockIdx.x;          // b*NWIN + w
    const int b    = bw >> 7;
    const int w    = bw & (NWIN - 1);
    const int tid  = threadIdx.x;
    const int wave = tid >> 5, lane = tid & 31;
    const int ln   = lane & 15, hi = lane >> 4;
    const long long rowBase = (long long)b * NTOK;

    __shared__ alignas(16) __bf16 kLds[WIN * CDIM];   // 32 KB K-window
    __shared__ alignas(16) float  sc[WIN * WIN];      // scores
    __shared__ alignas(16) __bf16 pb[WIN * WIN];      // probs (bf16)

    // Stage K window rows (shifted) into LDS: 32 rows x 64 uint4 chunks.
    for (int idx = tid; idx < WIN * (CDIM / 8); idx += 128) {
        int r  = idx >> 6;
        int ch = idx & 63;
        int tok = (w * WIN + r + SHIFT) & (NTOK - 1);
        stage16(kn + (rowBase + tok) * CDIM + ch * 8, kLds + r * CDIM + ch * 8);
    }
    async_join();
    __syncthreads();

    // ---- scores = Qw @ Kw^T : 4 waves <-> 4 tiles (mt, ntS) ----
    {
        const int mt = wave >> 1, ntS = wave & 1;
        const int qtok = (w * WIN + mt * 16 + ln + SHIFT) & (NTOK - 1);
        const __bf16* qbase = qn + (rowBase + qtok) * CDIM;      // A rows from global
        const __bf16* kbase = kLds + (ntS * 16 + ln) * CDIM;     // B cols = K rows
        v8f acc = zero8();
        for (int k0 = 0; k0 < CDIM; k0 += 32) {
            v16bf af = load_frag_a(qbase + k0, hi);
            v16bf bf = load_frag_bk(kbase + k0, hi);
            acc = wmma_bf16(af, bf, acc);
        }
#pragma unroll
        for (int r = 0; r < 8; ++r)
            sc[(mt * 16 + r + 8 * hi) * WIN + ntS * 16 + ln] = acc[r];
    }
    __syncthreads();

    // ---- softmax over each of the 32 rows (+ window mask) ----
    if (tid < WIN) {
        const float* mrow = mask + ((long long)w * WIN + tid) * WIN;
        float v[WIN];
        float mx = -3.4e38f;
#pragma unroll
        for (int j = 0; j < WIN; ++j) {
            v[j] = sc[tid * WIN + j] + mrow[j];
            mx = fmaxf(mx, v[j]);
        }
        float sum = 0.0f;
#pragma unroll
        for (int j = 0; j < WIN; ++j) { v[j] = expf(v[j] - mx); sum += v[j]; }
        const float inv = 1.0f / sum;
#pragma unroll
        for (int j = 0; j < WIN; ++j) pb[tid * WIN + j] = (__bf16)(v[j] * inv);
    }
    __syncthreads();

    // ---- x = probs @ Kw, fused scramble + roll + residual store ----
    // x[i,c] -> kv_out token p = c/16, channel c2 = (c%16)*32 + i.
    for (int idx = wave; idx < 64; idx += 4) {
        const int mt = idx & 1;      // row-halves of x
        const int nt = idx >> 1;     // 16-col group == destination token-in-window
        v16bf af = load_frag_a(pb + (mt * 16 + ln) * WIN, hi);
        Frag16 fb;
        const int col = nt * 16 + ln;
        const int kb  = hi ? 16 : 0;
#pragma unroll
        for (int kk = 0; kk < 16; ++kk)
            fb.e[kk] = kLds[(kb + kk) * CDIM + col];
        v8f acc = wmma_bf16(af, fb.v, zero8());

        const int tok = (w * WIN + nt + SHIFT) & (NTOK - 1);
        const long long base = (rowBase + tok) * CDIM + ln * 32 + mt * 16 + 8 * hi;
        float4 k0 = *(const float4*)(kv + base);
        float4 k1 = *(const float4*)(kv + base + 4);
        float4 o0, o1;
        o0.x = k0.x + acc[0]; o0.y = k0.y + acc[1];
        o0.z = k0.z + acc[2]; o0.w = k0.w + acc[3];
        o1.x = k1.x + acc[4]; o1.y = k1.y + acc[5];
        o1.z = k1.z + acc[6]; o1.w = k1.w + acc[7];
        *(float4*)(kv_out + base)     = o0;
        *(float4*)(kv_out + base + 4) = o1;
    }
}

// ---------------------------------------------------------------------------
// bf16 GEMM: out[M,N] = A[M,K] @ Bw[N,K]^T (+epilogue)
// Block tile 128x128x32, 256 threads (8 waves), wave tile 64x32.
// EPI 0: +bias, exact GELU, bf16 store.  EPI 1: +bias +residual, f32 store.
// ---------------------------------------------------------------------------
template<int EPI>
__global__ __launch_bounds__(256)
void gemm_bf16_kernel(const __bf16* __restrict__ A, const __bf16* __restrict__ Bw,
                      int K, __bf16* __restrict__ outB, float* __restrict__ outF,
                      const float* __restrict__ bias, const float* __restrict__ resid,
                      int N) {
    const int tid  = threadIdx.x;
    const int wave = tid >> 5, lane = tid & 31;
    const int ln   = lane & 15, hi = lane >> 4;
    const int mw   = wave >> 2, nw = wave & 3;   // 2x4 wave grid
    const int bm   = blockIdx.y * 128;
    const int bn   = blockIdx.x * 128;

    __shared__ alignas(16) __bf16 sA[128 * 32];
    __shared__ alignas(16) __bf16 sB[128 * 32];

    v8f acc[4][2];
#pragma unroll
    for (int mi = 0; mi < 4; ++mi)
#pragma unroll
        for (int ni = 0; ni < 2; ++ni) acc[mi][ni] = zero8();

    for (int k0 = 0; k0 < K; k0 += 32) {
        for (int idx = tid; idx < 512; idx += 256) {
            int r  = idx >> 2;
            int ch = idx & 3;
            const __bf16* ga = A  + (long long)(bm + r) * K + k0 + ch * 8;
            const __bf16* gb = Bw + (long long)(bn + r) * K + k0 + ch * 8;
            stage16(ga, sA + r * 32 + ch * 8);
            stage16(gb, sB + r * 32 + ch * 8);
            if (k0 + 32 < K) {                 // gfx1250 global_prefetch_b8
                __builtin_prefetch(ga + 32, 0, 1);
                __builtin_prefetch(gb + 32, 0, 1);
            }
        }
        async_join();
        __syncthreads();

        v16bf bf0 = load_frag_bk(sB + (nw * 32 + ln) * 32, hi);
        v16bf bf1 = load_frag_bk(sB + (nw * 32 + 16 + ln) * 32, hi);
#pragma unroll
        for (int mi = 0; mi < 4; ++mi) {
            v16bf af = load_frag_a(sA + (mw * 64 + mi * 16 + ln) * 32, hi);
            acc[mi][0] = wmma_bf16(af, bf0, acc[mi][0]);
            acc[mi][1] = wmma_bf16(af, bf1, acc[mi][1]);
        }
        __syncthreads();
    }

#pragma unroll
    for (int mi = 0; mi < 4; ++mi) {
#pragma unroll
        for (int ni = 0; ni < 2; ++ni) {
            const int n = bn + nw * 32 + ni * 16 + ln;
#pragma unroll
            for (int r = 0; r < 8; ++r) {
                const int m = bm + mw * 64 + mi * 16 + r + 8 * hi;
                const long long o = (long long)m * N + n;
                float v = acc[mi][ni][r] + bias[n];
                if (EPI == 0) {
                    outB[o] = (__bf16)gelu_exact(v);
                } else {
                    outF[o] = resid[o] + v;
                }
            }
        }
    }
}

// ---------------------------------------------------------------------------
// Host-side launch
// ---------------------------------------------------------------------------
extern "C" void kernel_launch(void* const* d_in, const int* in_sizes, int n_in,
                              void* d_out, int out_size, void* d_ws, size_t ws_size,
                              hipStream_t stream) {
    (void)in_sizes; (void)n_in; (void)out_size; (void)ws_size;
    const float* q     = (const float*)d_in[0];
    const float* kv    = (const float*)d_in[1];
    const float* mask  = (const float*)d_in[2];
    const float* n1w   = (const float*)d_in[3];
    const float* n1b   = (const float*)d_in[4];
    const float* n2w   = (const float*)d_in[5];
    const float* n2b   = (const float*)d_in[6];
    const float* fc1w  = (const float*)d_in[7];
    const float* fc1b  = (const float*)d_in[8];
    const float* fc2w  = (const float*)d_in[9];
    const float* fc2b  = (const float*)d_in[10];
    float* out = (float*)d_out;
    char*  ws  = (char*)d_ws;

    // Workspace layout (phase-disjoint aliasing of qn/kn with h1):
    //   [0, 134217728)            : h1 bf16 (32768x2048)  OR  qn(32MB)+kn(32MB)
    //   [134217728, 201326592)    : kv_out f32 (32768x512)
    //   [201326592, 234881024)    : act bf16 (32768x512)
    //   [234881024, 236978176)    : fc1_w bf16
    //   [236978176, 239075328)    : fc2_w bf16
    __bf16* qn  = (__bf16*)(ws);
    __bf16* kn  = (__bf16*)(ws + 33554432);
    __bf16* h1  = (__bf16*)(ws);
    float*  kvo = (float*) (ws + 134217728);
    __bf16* act = (__bf16*)(ws + 201326592);
    __bf16* w1  = (__bf16*)(ws + 234881024);
    __bf16* w2  = (__bf16*)(ws + 236978176);

    const int nw1 = HDIM * CDIM;   // 1048576
    cvt_bf16_kernel<<<(nw1 + 255) / 256, 256, 0, stream>>>(fc1w, w1, nw1);
    cvt_bf16_kernel<<<(nw1 + 255) / 256, 256, 0, stream>>>(fc2w, w2, nw1);

    layernorm_bf16_kernel<<<MTOK, 128, 0, stream>>>(q,  qn, n1w, n1b);
    layernorm_bf16_kernel<<<MTOK, 128, 0, stream>>>(kv, kn, n1w, n1b);

    attn_kernel<<<BDIM * NWIN, 128, 0, stream>>>(qn, kn, kv, mask, kvo);

    layernorm_bf16_kernel<<<MTOK, 128, 0, stream>>>(kvo, act, n2w, n2b);

    gemm_bf16_kernel<0><<<dim3(HDIM / 128, MTOK / 128), 256, 0, stream>>>(
        act, w1, CDIM, h1, nullptr, fc1b, nullptr, HDIM);
    gemm_bf16_kernel<1><<<dim3(CDIM / 128, MTOK / 128), 256, 0, stream>>>(
        h1, w2, HDIM, nullptr, out, fc2b, kvo, CDIM);
}
